// ConvolutionAutoEncoder_16037407883913
// MI455X (gfx1250) — compile-verified, matrix operand order checked by reading, not run on hardware
//
#include <hip/hip_runtime.h>
#include <math.h>

// ---------------------------------------------------------------------------
// Plain ext-vector types (trivial, union-safe; avoid HIP's class vector types)
// ---------------------------------------------------------------------------
typedef __attribute__((ext_vector_type(16))) __bf16        v16bf;
typedef __attribute__((ext_vector_type(8)))  float         v8f;
typedef __attribute__((ext_vector_type(4)))  unsigned int  v4u;
typedef __attribute__((ext_vector_type(2)))  unsigned int  v2u;
typedef __attribute__((ext_vector_type(4)))  float         v4f;

union BF16Frag { v4u q[2]; unsigned short u[16]; v16bf v; };

__device__ __forceinline__ unsigned short f32_to_bf16(float f) {
  unsigned int x = __float_as_uint(f);
  return (unsigned short)((x + 0x7FFFu + ((x >> 16) & 1u)) >> 16);  // RNE
}

__device__ __forceinline__ v8f wmma_bf16(const BF16Frag& a, const BF16Frag& b,
                                         v8f c) {
  return __builtin_amdgcn_wmma_f32_16x16x32_bf16(
      false, a.v, false, b.v, (short)0, c, false, false);
}

// ---------------------------------------------------------------------------
// conv1d (TF 'same') + tanh as implicit-im2col GEMM on v_wmma_f32_16x16x32_bf16.
//
// Block = 256 threads (8 waves) = 128 output rows (one batch, Lout % 128 == 0).
// A staged in LDS as bf16; per 32-deep K chunk each lane issues 2x ds_load_b128
// (the two 8-elem runs of the ISA 16-bit A layout:
//    K(lane,e) = ((e&8)<<1) + ((lane>>4)<<3) + (e&7), M = lane&15).
// B pre-packed in fragment order -> 2x global b128 per lane per N tile.
//
// NT (1/2/4 N-tiles, i.e. Cout 16/32/64) and IM2COL (Ktot<=32 staging mode)
// are template params so the K loop has NO runtime control flow: accumulators
// stay pinned in their WMMA register groups (no cross-branch v_mov copies).
// ---------------------------------------------------------------------------
template <int NT, bool IM2COL>
__global__ __launch_bounds__(256)
void conv1d_wmma_kernel(const float* __restrict__ x,             // [B, Lin, Cin]
                        const unsigned short* __restrict__ wfrag,// packed B frags
                        const float* __restrict__ bias,          // [Cout]
                        float* __restrict__ y,                   // [B, Lout, Cout]
                        int Lin, int Cin, int Ktap, int Cout,
                        int stride, int Lout, int pad)
{
  __shared__ unsigned short lds[16896];   // 33 KB (max: 132 rows x 128 ch)

  const int tid  = threadIdx.x;
  const int lane = tid & 31;
  const int wave = tid >> 5;
  const int half = lane >> 4;
  const int l16  = lane & 15;
  const int Ktot = Ktap * Cin;
  const int nChunks = (Ktot + 31) >> 5;

  const long mBase = (long)blockIdx.x * 128;
  const int  b     = (int)(mBase / Lout);  // Lout % 128 == 0 -> single batch
  const int  t0    = (int)(mBase % Lout);
  const int  tin0  = t0 * stride - pad;
  const float* xb  = x + (long)b * Lin * Cin;

  // ---------------- stage A tile into LDS (bf16) ----------------
  if (IM2COL) {
    // lds[mloc][32] = zero-padded im2col chunk (handles any Cin)
    for (int i = tid; i < 128 * 32; i += 256) {
      const int mloc = i >> 5, k = i & 31;
      unsigned short v = 0;
      if (k < Ktot) {
        const int tap = k / Cin;
        const int cin = k - tap * Cin;
        const int ti  = tin0 + mloc * stride + tap;
        if (ti >= 0 && ti < Lin) v = f32_to_bf16(xb[(long)ti * Cin + cin]);
      }
      lds[i] = v;
    }
  } else {
    // raw input rows; requires Cin % 32 == 0 (all heavy layers: 64 or 128)
    const int rows   = 128 * stride + Ktap - 1;
    const int cshift = 31 - __clz(Cin);          // Cin is a power of two here
    const int total4 = (rows * Cin) >> 2;
    for (int i = tid; i < total4; i += 256) {
      const int idx = i << 2;
      const int r   = idx >> cshift;
      const int c   = idx & (Cin - 1);
      const int ti  = tin0 + r;
      v2u p = {0u, 0u};
      if (ti >= 0 && ti < Lin) {
        const v4f f = *(const v4f*)(xb + (long)ti * Cin + c);   // 16B coalesced
        p.x = ((unsigned)f32_to_bf16(f[1]) << 16) | f32_to_bf16(f[0]);
        p.y = ((unsigned)f32_to_bf16(f[3]) << 16) | f32_to_bf16(f[2]);
      }
      *(v2u*)(lds + idx) = p;
    }
  }
  __syncthreads();

  // ---------------- GEMM main loop (branch-free) ----------------
  v8f acc0 = {}, acc1 = {}, acc2 = {}, acc3 = {};
  const int mloc = wave * 16 + l16;

  for (int kc = 0; kc < nChunks; ++kc) {
    // A fragment: two contiguous 8-elem bf16 runs from LDS
    const unsigned short* ap;
    if (IM2COL) {
      ap = lds + mloc * 32 + half * 8;
    } else {
      const int kb  = kc << 5;
      const int tap = kb / Cin;              // constant across the chunk
      const int cb  = kb & (Cin - 1);
      ap = lds + (mloc * stride + tap) * Cin + cb + half * 8;
    }
    BF16Frag a;
    a.q[0] = *(const v4u*)(ap);        // ds_load_b128
    a.q[1] = *(const v4u*)(ap + 16);   // ds_load_b128

    // B fragments: 512 ushorts (= one 32-lane fragment) per N tile
    const unsigned short* wp = wfrag + ((long)kc * NT * 32 + lane) * 16;
    BF16Frag b0, b1, b2, b3;
    b0.q[0] = *(const v4u*)(wp);
    b0.q[1] = *(const v4u*)(wp + 8);
    if (NT > 1) {
      b1.q[0] = *(const v4u*)(wp + 512);
      b1.q[1] = *(const v4u*)(wp + 520);
    }
    if (NT > 2) {
      b2.q[0] = *(const v4u*)(wp + 1024);
      b2.q[1] = *(const v4u*)(wp + 1032);
      b3.q[0] = *(const v4u*)(wp + 1536);
      b3.q[1] = *(const v4u*)(wp + 1544);
    }

    acc0 = wmma_bf16(a, b0, acc0);
    if (NT > 1) acc1 = wmma_bf16(a, b1, acc1);
    if (NT > 2) {
      acc2 = wmma_bf16(a, b2, acc2);
      acc3 = wmma_bf16(a, b3, acc3);
    }
  }

  // ---------------- epilogue: bias + tanh ----------------
  const int trow0 = t0 + wave * 16;
  float* yb = y + (long)b * Lout * Cout;
  auto store_tile = [&](const v8f& acc, int nt) {
    const int n = nt * 16 + l16;
    if (n < Cout) {
      const float bv = bias[n];
#pragma unroll
      for (int r = 0; r < 8; ++r) {
        const int tt = trow0 + (half ? r + 8 : r);
        yb[(long)tt * Cout + n] = tanhf(acc[r] + bv);
      }
    }
  };
  store_tile(acc0, 0);
  if (NT > 1) store_tile(acc1, 1);
  if (NT > 2) { store_tile(acc2, 2); store_tile(acc3, 3); }
}

// ---------------------------------------------------------------------------
// Weight prep: f32 [Ktot, Cout] -> bf16 WMMA B-fragments
//   out[(((kc*nTiles + nt)*32 + lane)*16 + e)], K(lane,e) per ISA layout.
// ---------------------------------------------------------------------------
__global__ void pack_wfrag_kernel(const float* __restrict__ w,
                                  unsigned short* __restrict__ out,
                                  int Ktot, int Cout, int nTiles, int nChunks) {
  const int i = blockIdx.x * blockDim.x + threadIdx.x;
  const int total = nChunks * nTiles * 512;
  if (i >= total) return;
  const int e    = i & 15;
  const int lane = (i >> 4) & 31;
  const int nt   = (i >> 9) % nTiles;
  const int kc   = (i >> 9) / nTiles;
  const int half = lane >> 4, l16 = lane & 15;
  const int k = kc * 32 + ((e & 8) << 1) + (half << 3) + (e & 7);
  const int n = nt * 16 + l16;
  unsigned short v = 0;
  if (k < Ktot && n < Cout) v = f32_to_bf16(w[(long)k * Cout + n]);
  out[i] = v;
}

// ---------------------------------------------------------------------------
// inp seed: impulse at t==0 over [B, L, C]
// ---------------------------------------------------------------------------
__global__ void impulse_kernel(float* __restrict__ out, int L, int C, long total) {
  long i = (long)blockIdx.x * blockDim.x + threadIdx.x;
  if (i >= total) return;
  long t = (i / C) % L;
  out[i] = (t == 0) ? 1.0f : 0.0f;
}

// ---------------------------------------------------------------------------
// channel concat: [rows,Ca] ++ [rows,Cb] -> [rows,Ca+Cb]
// ---------------------------------------------------------------------------
__global__ void concat_kernel(const float* __restrict__ a, const float* __restrict__ b,
                              float* __restrict__ out, long rows, int Ca, int Cb) {
  const int Cc = Ca + Cb;
  long i = (long)blockIdx.x * blockDim.x + threadIdx.x;
  long total = rows * Cc;
  if (i >= total) return;
  long r = i / Cc;
  int  c = (int)(i - r * Cc);
  out[i] = (c < Ca) ? a[r * Ca + c] : b[r * Cb + (c - Ca)];
}

// ---------------------------------------------------------------------------
// Per-(batch,channel) circular convolution, length 8192, filter support 4096.
// Packed-real trick: z = x + i*f, forward complex FFT, Hermitian split +
// spectral multiply, inverse complex FFT; in 64KB LDS (<< 320KB/WGP).
// ---------------------------------------------------------------------------
#define FFT_N   8192
#define FFT_LOG 13

__device__ void fft_inplace(float2* z, int tid, int nth, float sign) {
  for (int i = tid; i < FFT_N; i += nth) {
    int j = (int)(__brev((unsigned)i) >> (32 - FFT_LOG));
    if (i < j) { float2 tmp = z[i]; z[i] = z[j]; z[j] = tmp; }
  }
  __syncthreads();
  for (int s = 1; s <= FFT_LOG; ++s) {
    const int halfm = 1 << (s - 1);
    const float angStep = sign * 6.2831853071795864769f / (float)(1 << s);
    for (int bf = tid; bf < FFT_N / 2; bf += nth) {
      const int pos = bf & (halfm - 1);
      const int grp = bf >> (s - 1);
      const int i0  = (grp << s) + pos;
      const int i1  = i0 + halfm;
      const float ang = angStep * (float)pos;
      const float wr = __cosf(ang), wi = __sinf(ang);
      const float2 u = z[i0], v = z[i1];
      const float tr = v.x * wr - v.y * wi;
      const float ti = v.x * wi + v.y * wr;
      z[i0] = make_float2(u.x + tr, u.y + ti);
      z[i1] = make_float2(u.x - tr, u.y - ti);
    }
    __syncthreads();
  }
}

__global__ __launch_bounds__(256)
void circconv_fft_kernel(const float* __restrict__ xin,   // [B, N, C]
                         const float* __restrict__ filt,  // [B, N/2, C]
                         float* __restrict__ yout,        // [B, N, C]
                         int C)
{
  __shared__ float2 z[FFT_N];
  const int bc = blockIdx.x;
  const int b = bc / C, c = bc % C;
  const int tid = threadIdx.x, nth = blockDim.x;
  const int Lf = FFT_N / 2;

  for (int t = tid; t < FFT_N; t += nth) {
    const float re = xin[((long)b * FFT_N + t) * C + c];
    const float im = (t < Lf) ? filt[((long)b * Lf + t) * C + c] : 0.0f;
    z[t] = make_float2(re, im);
  }
  __syncthreads();

  fft_inplace(z, tid, nth, -1.0f);   // forward

  for (int k = tid; k <= FFT_N / 2; k += nth) {
    if (k == 0 || k == FFT_N / 2) {
      const float2 zk = z[k];
      z[k] = make_float2(zk.x * zk.y, 0.0f);
    } else {
      const float2 zk = z[k], zn = z[FFT_N - k];
      const float Xr = 0.5f * (zk.x + zn.x);
      const float Xi = 0.5f * (zk.y - zn.y);
      const float Fr = 0.5f * (zk.y + zn.y);
      const float Fi = -0.5f * (zk.x - zn.x);
      const float Yr = Xr * Fr - Xi * Fi;
      const float Yi = Xr * Fi + Xi * Fr;
      z[k]         = make_float2(Yr,  Yi);
      z[FFT_N - k] = make_float2(Yr, -Yi);
    }
  }
  __syncthreads();

  fft_inplace(z, tid, nth, +1.0f);   // inverse (unscaled)

  const float scale = 1.0f / (float)FFT_N;
  for (int t = tid; t < FFT_N; t += nth)
    yout[((long)b * FFT_N + t) * C + c] = z[t].x * scale;
}

// ---------------------------------------------------------------------------
// Host orchestration.  Input flattening (setup_inputs insertion order):
//   [0] x ; enc [1..6] ; fgb [7..46] ; cb [47..54] ; out [55..60]
// ---------------------------------------------------------------------------
extern "C" void kernel_launch(void* const* d_in, const int* in_sizes, int n_in,
                              void* d_out, int out_size, void* d_ws, size_t ws_size,
                              hipStream_t stream)
{
  (void)in_sizes; (void)n_in; (void)out_size; (void)ws_size;
  const int B = 32, L = 8192, C = 64, Lh = 4096;

  // ---- carve workspace ----
  char* ws = (char*)d_ws;
  size_t off = 0;
  auto alloc_f = [&](size_t elems) -> float* {
    float* p = (float*)(ws + off);
    off = (off + elems * sizeof(float) + 255) & ~(size_t)255;
    return p;
  };
  float* bufA = alloc_f((size_t)B * L * C);        // inp ping      (64 MB)
  float* bufB = alloc_f((size_t)B * L * C);        // inp pong      (64 MB)
  float* feat = alloc_f((size_t)B * Lh * C);       // encoder out   (32 MB)
  float* filt = alloc_f((size_t)B * Lh * C);       // filter        (32 MB)
  float* cat  = alloc_f((size_t)B * Lh * 2 * C);   // concat        (64 MB)
  float* t0   = alloc_f((size_t)B * Lh * C);       // fgb ping      (32 MB)
  float* t1   = alloc_f((size_t)B * Lh * C);       // fgb pong      (32 MB)
  unsigned short* wheap = (unsigned short*)(ws + off);
  size_t woff = 0;

  auto prep = [&](int idx, int Ktot, int Cout) -> const unsigned short* {
    const int nTiles  = (Cout + 15) >> 4;
    const int nChunks = (Ktot + 31) >> 5;
    const int elems   = nChunks * nTiles * 512;
    unsigned short* dst = wheap + woff;
    woff += ((size_t)elems + 127) & ~(size_t)127;
    pack_wfrag_kernel<<<(elems + 255) / 256, 256, 0, stream>>>(
        (const float*)d_in[idx], dst, Ktot, Cout, nTiles, nChunks);
    return dst;
  };
  auto conv = [&](const float* in, const unsigned short* wq, int bidx, float* out,
                  int Lin, int Cin, int K, int Cout, int stride) {
    const int Lout = (Lin + stride - 1) / stride;        // TF 'same'
    int padTotal = (Lout - 1) * stride + K - Lin;
    if (padTotal < 0) padTotal = 0;
    const int pad  = padTotal / 2;
    const int grid = (int)(((long)B * Lout) / 128);
    const bool im2col = (K * Cin <= 32);
    const int  nTiles = (Cout + 15) >> 4;
    const float* bias = (const float*)d_in[bidx];
    if (im2col) {
      if (nTiles >= 4)
        conv1d_wmma_kernel<4, true><<<grid, 256, 0, stream>>>(
            in, wq, bias, out, Lin, Cin, K, Cout, stride, Lout, pad);
      else if (nTiles == 2)
        conv1d_wmma_kernel<2, true><<<grid, 256, 0, stream>>>(
            in, wq, bias, out, Lin, Cin, K, Cout, stride, Lout, pad);
      else
        conv1d_wmma_kernel<1, true><<<grid, 256, 0, stream>>>(
            in, wq, bias, out, Lin, Cin, K, Cout, stride, Lout, pad);
    } else {
      if (nTiles >= 4)
        conv1d_wmma_kernel<4, false><<<grid, 256, 0, stream>>>(
            in, wq, bias, out, Lin, Cin, K, Cout, stride, Lout, pad);
      else if (nTiles == 2)
        conv1d_wmma_kernel<2, false><<<grid, 256, 0, stream>>>(
            in, wq, bias, out, Lin, Cin, K, Cout, stride, Lout, pad);
      else
        conv1d_wmma_kernel<1, false><<<grid, 256, 0, stream>>>(
            in, wq, bias, out, Lin, Cin, K, Cout, stride, Lout, pad);
    }
  };

  // ---- encoder ----
  conv((const float*)d_in[0], prep(1, 5 * 1,  64), 2, bufA, L, 1,  5, 64, 1);
  conv(bufA,                  prep(3, 3 * 64, 64), 4, bufB, L, 64, 3, 64, 1);
  conv(bufB,                  prep(5, 3 * 64, 64), 6, feat, L, 64, 3, 64, 2);

  // ---- seed impulse ----
  {
    long total = (long)B * L * C;
    impulse_kernel<<<(int)((total + 255) / 256), 256, 0, stream>>>(bufA, L, C, total);
  }

  // ---- 4 filter-generation + circular-conv rounds ----
  for (int i = 0; i < 4; ++i) {
    const float* fin;
    int cin0;
    if (i == 0) { fin = feat; cin0 = 64; }
    else {
      const long rows = (long)B * Lh;
      const long total = rows * 128;
      concat_kernel<<<(int)((total + 255) / 256), 256, 0, stream>>>(
          feat, filt, cat, rows, 64, 64);
      fin = cat; cin0 = 128;
    }
    const int base = 7 + i * 10;
    conv(fin, prep(base + 0, 5 * cin0, 64), base + 1, t0,   Lh, cin0, 5, 64, 1);
    conv(t0,  prep(base + 2, 5 * 64,   64), base + 3, t1,   Lh, 64,   5, 64, 1);
    conv(t1,  prep(base + 4, 5 * 64,   64), base + 5, t0,   Lh, 64,   5, 64, 1);
    conv(t0,  prep(base + 6, 5 * 64,   64), base + 7, t1,   Lh, 64,   5, 64, 1);
    conv(t1,  prep(base + 8, 5 * 64,   64), base + 9, filt, Lh, 64,   5, 64, 1);

    circconv_fft_kernel<<<B * C, 256, 0, stream>>>(bufA, filt, bufB, C);

    const int cbase = 47 + i * 2;
    conv(bufB, prep(cbase, 5 * 64, 64), cbase + 1, bufA, L, 64, 5, 64, 1);
  }

  // ---- output head ----
  conv(bufA, prep(55, 5 * 64, 64), 56, bufB,          L, 64, 5, 64, 1);
  conv(bufB, prep(57, 3 * 64, 32), 58, t0,            L, 64, 3, 32, 1);
  conv(t0,   prep(59, 1 * 32, 1),  60, (float*)d_out, L, 32, 1, 1,  1);
}
